// InterpretableMultiHeadAttention_8074538516732
// MI455X (gfx1250) — compile-verified, matrix operand order checked by reading, not run on hardware
//
#include <hip/hip_runtime.h>
#include <hip/hip_bf16.h>

// ---------------------------------------------------------------------------
// InterpretableMultiHeadAttention for MI455X (gfx1250, wave32, WMMA)
// B=4, S=2048, D=256, H=4, DH=64
// d_out = out [B,S,D] f32  ||  attn [B,H,S,S] f32
// ---------------------------------------------------------------------------

#define BB 4
#define SS 2048
#define DD 256
#define HH 4
#define DHH 64

typedef _Float16 half_t;
typedef __attribute__((ext_vector_type(16))) _Float16 v16h;
typedef __attribute__((ext_vector_type(8)))  _Float16 v8h;
typedef __attribute__((ext_vector_type(8)))  float    v8f;

#define WMMA_F16(a, b, c) \
  __builtin_amdgcn_wmma_f32_16x16x32_f16(false, (a), false, (b), (short)0, (c), false, false)

__device__ __forceinline__ v16h concat8(v8h lo, v8h hi) {
  return __builtin_shufflevector(lo, hi, 0,1,2,3,4,5,6,7,8,9,10,11,12,13,14,15);
}

// CDNA5 16-bit A-matrix 16x32 fragment from row-major f16 (also used as the
// B-fragment for NT GEMM: load rows of B^T).  ISA layout: lanes 0-15 hold
// M=lane, K = {0..7, 16..23}; lanes 16-31 hold M=lane-16, K = {8..15, 24..31}.
__device__ __forceinline__ v16h load_frag_f16(const half_t* base, int ld) {
  int lane = threadIdx.x & 31;
  int m    = lane & 15;
  int klo  = (lane >> 4) << 3;
  const half_t* p = base + (size_t)m * ld + klo;
  v8h lo = *(const v8h*)p;
  v8h hi = *(const v8h*)(p + 16);
  return concat8(lo, hi);
}

// Same fragment, source f32 converted to f16 on load.
__device__ __forceinline__ v16h load_frag_f32(const float* base, int ld) {
  int lane = threadIdx.x & 31;
  int m    = lane & 15;
  int klo  = (lane >> 4) << 3;
  const float* p = base + (size_t)m * ld + klo;
  v8h lo = {}, hi = {};
#pragma unroll
  for (int i = 0; i < 8; ++i) { lo[i] = (half_t)p[i]; hi[i] = (half_t)p[16 + i]; }
  return concat8(lo, hi);
}

// ---------------------------------------------------------------------------
// Kernel 0: weight transpose + f32->f16 so all B operands load row-major.
// WqT/WkT [H][DH][D], WvT [DH][D], WfT [D][DH]   (all = W^T per head)
// ---------------------------------------------------------------------------
__global__ void imha_prep_weights(const float* __restrict__ Wq, const float* __restrict__ Wk,
                                  const float* __restrict__ Wv, const float* __restrict__ Wf,
                                  half_t* __restrict__ WqT, half_t* __restrict__ WkT,
                                  half_t* __restrict__ WvT, half_t* __restrict__ WfT) {
  int tid = blockIdx.x * blockDim.x + threadIdx.x;
  int nt  = gridDim.x * blockDim.x;
  for (int i = tid; i < HH * DHH * DD; i += nt) {
    int h = i / (DHH * DD); int r = i % (DHH * DD);
    int e = r / DD;         int d = r % DD;
    WqT[i] = (half_t)Wq[((size_t)h * DD + d) * DHH + e];
    WkT[i] = (half_t)Wk[((size_t)h * DD + d) * DHH + e];
  }
  for (int i = tid; i < DHH * DD; i += nt) {
    int e = i / DD, d = i % DD;
    WvT[i] = (half_t)Wv[(size_t)d * DHH + e];
  }
  for (int i = tid; i < DD * DHH; i += nt) {
    int n = i / DHH, kk = i % DHH;
    WfT[i] = (half_t)Wf[(size_t)kk * DD + n];
  }
}

// ---------------------------------------------------------------------------
// Kernel 1: q_h / k_h projection.  grid (S/16, B*H, 2), block 256 (8 waves).
// Each block: 16 rows x 64 cols, K=256.  Wave w: n-tile (w&3), K-half (w>>2).
// ---------------------------------------------------------------------------
__global__ void imha_proj_qk(const float* __restrict__ q, const float* __restrict__ k,
                             const half_t* __restrict__ WqT, const half_t* __restrict__ WkT,
                             const float* __restrict__ bq, const float* __restrict__ bk,
                             half_t* __restrict__ qh, half_t* __restrict__ kh) {
  __shared__ float red[8 * 256];
  int s0 = blockIdx.x * 16;
  int b  = blockIdx.y >> 2;
  int h  = blockIdx.y & 3;
  int wave = threadIdx.x >> 5;
  int lane = threadIdx.x & 31;

  const float*  X    = (blockIdx.z == 0) ? q   : k;
  const half_t* WT   = (blockIdx.z == 0) ? WqT : WkT;
  const float*  bias = (blockIdx.z == 0) ? bq  : bk;
  half_t*       Y    = (blockIdx.z == 0) ? qh  : kh;

  int e0  = (wave & 3) * 16;
  int khf = wave >> 2;
  const float*  abase = X  + ((size_t)b * SS + s0) * DD;
  const half_t* bbase = WT + ((size_t)h * DHH + e0) * DD;

  v8f c = {};
#pragma unroll
  for (int j = 0; j < 4; ++j) {
    int k0 = khf * 128 + j * 32;
    v16h a  = load_frag_f32(abase + k0, DD);
    v16h bm = load_frag_f16(bbase + k0, DD);
    c = WMMA_F16(a, bm, c);
  }
#pragma unroll
  for (int r = 0; r < 8; ++r) red[wave * 256 + r * 32 + lane] = c[r];
  __syncthreads();
  if (wave < 4) {
    int n = lane & 15, hi = lane >> 4;
    half_t* outp = Y + (((size_t)(b * HH + h)) * SS + s0) * DHH + e0;
    float bval = bias[h * DHH + e0 + n];
#pragma unroll
    for (int r = 0; r < 8; ++r) {
      float val = red[wave * 256 + r * 32 + lane] + red[(wave + 4) * 256 + r * 32 + lane];
      int m = r + 8 * hi;
      outp[(size_t)m * DHH + n] = (half_t)(val + bval);
    }
  }
}

// ---------------------------------------------------------------------------
// Kernel 1v: v_p projection, stored transposed: vpT [B][DH][S] f16
// ---------------------------------------------------------------------------
__global__ void imha_proj_v(const float* __restrict__ v, const half_t* __restrict__ WvT,
                            const float* __restrict__ bv, half_t* __restrict__ vpT) {
  __shared__ float red[8 * 256];
  int t0 = blockIdx.x * 16;
  int b  = blockIdx.y;
  int wave = threadIdx.x >> 5;
  int lane = threadIdx.x & 31;
  int e0  = (wave & 3) * 16;
  int khf = wave >> 2;
  const float*  abase = v   + ((size_t)b * SS + t0) * DD;
  const half_t* bbase = WvT + (size_t)e0 * DD;

  v8f c = {};
#pragma unroll
  for (int j = 0; j < 4; ++j) {
    int k0 = khf * 128 + j * 32;
    v16h a  = load_frag_f32(abase + k0, DD);
    v16h bm = load_frag_f16(bbase + k0, DD);
    c = WMMA_F16(a, bm, c);
  }
#pragma unroll
  for (int r = 0; r < 8; ++r) red[wave * 256 + r * 32 + lane] = c[r];
  __syncthreads();
  if (wave < 4) {
    int n = lane & 15, hi = lane >> 4;
    half_t* outp = vpT + (size_t)b * DHH * SS;
    float bval = bv[e0 + n];
#pragma unroll
    for (int r = 0; r < 8; ++r) {
      float val = red[wave * 256 + r * 32 + lane] + red[(wave + 4) * 256 + r * 32 + lane] + bval;
      int m = r + 8 * hi;
      outp[(size_t)(e0 + n) * SS + (t0 + m)] = (half_t)val;  // transposed store
    }
  }
}

// ---------------------------------------------------------------------------
// Kernel 2: fused scores + mask + softmax + attn store + PV.
// grid (S/16, B*H), block 256.  Dynamic LDS holds a 16 x 2048 f32 score strip
// (row stride 2056 -> stride%64==8 banks, worst-case 2-way conflicts).
// ---------------------------------------------------------------------------
#define SROW 2056

__global__ void imha_attn(const half_t* __restrict__ qh, const half_t* __restrict__ kh,
                          const half_t* __restrict__ vpT, const unsigned char* __restrict__ mask,
                          float* __restrict__ attn, float* __restrict__ heads) {
  extern __shared__ float smem[];
  float* sc     = smem;               // [16][SROW]
  float* rowmax = smem + 16 * SROW;   // [16]
  float* rowinv = rowmax + 16;        // [16]
  float* red    = rowinv + 16;        // [16*16]
  float* part   = red + 256;          // [8*256]

  int s0 = blockIdx.x * 16;
  int b  = blockIdx.y >> 2;
  int h  = blockIdx.y & 3;
  int tid  = threadIdx.x;
  int wave = tid >> 5;
  int lane = tid & 31;
  int n   = lane & 15;
  int hig = lane >> 4;

  const half_t* qbase = qh + (((size_t)(b * HH + h)) * SS + s0) * DHH;
  const half_t* kbase = kh + (((size_t)(b * HH + h)) * SS) * DHH;
  const unsigned char* mbase = mask + ((size_t)b * SS + s0) * SS;

  // Q fragments (K = 0..31, 32..63), reused across all t-tiles.
  v16h aq0 = load_frag_f16(qbase + 0,  DHH);
  v16h aq1 = load_frag_f16(qbase + 32, DHH);

  // ---- Phase A: scores = (q_h k_h^T)/8, masked, into LDS ----
  for (int t0 = wave * 16; t0 < SS; t0 += 128) {
    v16h bk0 = load_frag_f16(kbase + (size_t)t0 * DHH + 0,  DHH);
    v16h bk1 = load_frag_f16(kbase + (size_t)t0 * DHH + 32, DHH);
    v8f c = {};
    c = WMMA_F16(aq0, bk0, c);
    c = WMMA_F16(aq1, bk1, c);
#pragma unroll
    for (int r = 0; r < 8; ++r) {
      int m = r + 8 * hig;
      float s = c[r] * 0.125f;                       // 1/sqrt(DH)
      if (mbase[(size_t)m * SS + (t0 + n)]) s = -1e9f;
      sc[m * SROW + t0 + n] = s;
    }
  }
  __syncthreads();

  // ---- Phase B: softmax stats (16 threads per row) ----
  {
    int row = tid >> 4, chunk = tid & 15;
    const float* rr = sc + row * SROW + chunk * 128;
    float mx = -3.4e38f;
    for (int i = 0; i < 128; ++i) mx = fmaxf(mx, rr[i]);
    red[row * 16 + chunk] = mx;
    __syncthreads();
    if (chunk == 0) {
      float m2 = -3.4e38f;
      for (int i = 0; i < 16; ++i) m2 = fmaxf(m2, red[row * 16 + i]);
      rowmax[row] = m2;
    }
    __syncthreads();
    float rm = rowmax[row];
    float* rw = sc + row * SROW + chunk * 128;
    float sum = 0.f;
    for (int i = 0; i < 128; ++i) { float e = __expf(rw[i] - rm); rw[i] = e; sum += e; }
    red[row * 16 + chunk] = sum;
    __syncthreads();
    if (chunk == 0) {
      float s2 = 0.f;
      for (int i = 0; i < 16; ++i) s2 += red[row * 16 + i];
      rowinv[row] = 1.0f / s2;
    }
    __syncthreads();
  }

  // ---- Phase C: attn store (coalesced) + PV GEMM out of LDS ----
  {
    float* ab = attn + (((size_t)(b * HH + h)) * SS + s0) * SS;
    for (int idx = tid; idx < 16 * SS; idx += 256) {
      int m = idx >> 11;          // / 2048
      int t = idx & (SS - 1);
      ab[(size_t)m * SS + t] = sc[m * SROW + t] * rowinv[m];
    }

    int e0  = (wave & 3) * 16;
    int khf = wave >> 2;
    const half_t* vb = vpT + ((size_t)b * DHH + e0) * SS;
    int mA   = lane & 15;
    int kloA = (lane >> 4) << 3;
    float invA = rowinv[mA];

    v8f c = {};
    for (int t0 = khf * 1024; t0 < khf * 1024 + 1024; t0 += 32) {
      const float* pa = sc + mA * SROW + t0 + kloA;
      v8h lo = {}, hi2 = {};
#pragma unroll
      for (int i = 0; i < 8; ++i) {
        lo[i]  = (half_t)(pa[i]      * invA);
        hi2[i] = (half_t)(pa[16 + i] * invA);
      }
      v16h a  = concat8(lo, hi2);
      v16h bm = load_frag_f16(vb + t0, SS);
      c = WMMA_F16(a, bm, c);
    }
    __syncthreads();                // all LDS score reads done
#pragma unroll
    for (int r = 0; r < 8; ++r) part[wave * 256 + r * 32 + lane] = c[r];
    __syncthreads();
    if (wave < 4) {
      float* hb = heads + (((size_t)(b * HH + h)) * SS + s0) * DHH + e0;
#pragma unroll
      for (int r = 0; r < 8; ++r) {
        float vsum = part[wave * 256 + r * 32 + lane] + part[(wave + 4) * 256 + r * 32 + lane];
        int m = r + 8 * hig;
        hb[(size_t)m * DHH + n] = vsum;
      }
    }
  }
}

// ---------------------------------------------------------------------------
// Kernel 3: mean over heads + final projection.  grid (S/16, B), block 256.
// ---------------------------------------------------------------------------
__global__ void imha_final(const float* __restrict__ heads, const half_t* __restrict__ WfT,
                           float* __restrict__ out) {
  __shared__ half_t mtile[16 * 72];   // 16 x 64 (+8 pad)
  int s0 = blockIdx.x * 16;
  int b  = blockIdx.y;
  int tid  = threadIdx.x;
  int wave = tid >> 5;
  int lane = tid & 31;

  for (int idx = tid; idx < 16 * DHH; idx += 256) {
    int m = idx >> 6, e = idx & 63;
    const float* hb = heads + ((size_t)b * HH * SS + (s0 + m)) * DHH + e;
    float s = 0.f;
#pragma unroll
    for (int h = 0; h < HH; ++h) s += hb[(size_t)h * SS * DHH];
    mtile[m * 72 + e] = (half_t)(s * 0.25f);
  }
  __syncthreads();

  v16h a0 = load_frag_f16(mtile + 0,  72);
  v16h a1 = load_frag_f16(mtile + 32, 72);
  int n = lane & 15, hig = lane >> 4;
#pragma unroll
  for (int i = 0; i < 2; ++i) {
    int n0 = (wave * 2 + i) * 16;
    const half_t* bb = WfT + (size_t)n0 * DHH;
    v16h b0 = load_frag_f16(bb + 0,  DHH);
    v16h b1 = load_frag_f16(bb + 32, DHH);
    v8f c = {};
    c = WMMA_F16(a0, b0, c);
    c = WMMA_F16(a1, b1, c);
    float* ob = out + ((size_t)b * SS + s0) * DD + n0;
#pragma unroll
    for (int r = 0; r < 8; ++r) {
      int m = r + 8 * hig;
      ob[(size_t)m * DD + n] = c[r];
    }
  }
}

// ---------------------------------------------------------------------------
extern "C" void kernel_launch(void* const* d_in, const int* in_sizes, int n_in,
                              void* d_out, int out_size, void* d_ws, size_t ws_size,
                              hipStream_t stream) {
  const float* q  = (const float*)d_in[0];
  const float* k  = (const float*)d_in[1];
  const float* v  = (const float*)d_in[2];
  const unsigned char* mask = (const unsigned char*)d_in[3];   // bool -> 1 byte
  const float* Wq = (const float*)d_in[4];
  const float* bq = (const float*)d_in[5];
  const float* Wk = (const float*)d_in[6];
  const float* bk = (const float*)d_in[7];
  const float* Wv = (const float*)d_in[8];
  const float* bv = (const float*)d_in[9];
  const float* Wf = (const float*)d_in[10];

  float* out  = (float*)d_out;                         // [B,S,D]
  float* attn = out + (size_t)BB * SS * DD;            // [B,H,S,S]

  // Workspace carve-up (~17.3 MB)
  char* ws = (char*)d_ws;
  auto take = [&](size_t bytes) { char* p = ws; ws += (bytes + 255) & ~(size_t)255; return p; };
  half_t* WqT  = (half_t*)take((size_t)HH * DHH * DD * sizeof(half_t));
  half_t* WkT  = (half_t*)take((size_t)HH * DHH * DD * sizeof(half_t));
  half_t* WvT  = (half_t*)take((size_t)DHH * DD * sizeof(half_t));
  half_t* WfT  = (half_t*)take((size_t)DD * DHH * sizeof(half_t));
  half_t* qhW  = (half_t*)take((size_t)BB * HH * SS * DHH * sizeof(half_t));
  half_t* khW  = (half_t*)take((size_t)BB * HH * SS * DHH * sizeof(half_t));
  half_t* vpT  = (half_t*)take((size_t)BB * DHH * SS * sizeof(half_t));
  float*  heads = (float*)take((size_t)BB * HH * SS * DHH * sizeof(float));

  imha_prep_weights<<<64, 256, 0, stream>>>(Wq, Wk, Wv, Wf, WqT, WkT, WvT, WfT);

  imha_proj_qk<<<dim3(SS / 16, BB * HH, 2), 256, 0, stream>>>(q, k, WqT, WkT, bq, bk, qhW, khW);
  imha_proj_v <<<dim3(SS / 16, BB), 256, 0, stream>>>(v, WvT, bv, vpT);

  // 16 rows x SROW f32 score strip + softmax scratch; ~138 KB of the 320 KB/WGP LDS.
  constexpr size_t ATTN_LDS = (size_t)(16 * SROW + 16 + 16 + 256 + 8 * 256) * sizeof(float);
  imha_attn<<<dim3(SS / 16, BB * HH), 256, ATTN_LDS, stream>>>(qhW, khW, vpT, mask, attn, heads);

  imha_final<<<dim3(SS / 16, BB), 256, 0, stream>>>(heads, WfT, out);
}
